// RealUpdateLayer2UU_32796370272297
// MI455X (gfx1250) — compile-verified, matrix operand order checked by reading, not run on hardware
//
#include <hip/hip_runtime.h>

// ---- problem constants (match reference) ----
#define Bb 16
#define Ll 128
#define Kk 64
#define Ee 256
#define Hh 8
#define Dd 32
#define ROWS (Bb * Ll)                  // 2048
#define ATT_SCALE 0.17677669529663687f  // 1/sqrt(32)

typedef __attribute__((ext_vector_type(16))) _Float16 v16h;
typedef __attribute__((ext_vector_type(8)))  _Float16 v8h;
typedef __attribute__((ext_vector_type(8)))  float    v8f;

__device__ __forceinline__ v8f vzero8() {
  v8f z;
#pragma unroll
  for (int i = 0; i < 8; ++i) z[i] = 0.0f;
  return z;
}

// WMMA A/B fragment load from row-major f16 LDS tile.
// p = tile + row*stride + 8*kh; elems 0..7 = cols [8kh,8kh+8),
// elems 8..15 = cols [16+8kh,16+8kh+8) (CDNA5 ISA 16-bit A layout).
__device__ __forceinline__ v16h frag_ld(const _Float16* p) {
  union { v16h v; v8h h[2]; } u;
  u.h[0] = *(const v8h*)(p);
  u.h[1] = *(const v8h*)(p + 16);
  return u.v;
}

__device__ __forceinline__ v8f wmma16(v16h a, v16h b, v8f c) {
  return __builtin_amdgcn_wmma_f32_16x16x32_f16(false, a, false, b, (short)0, c,
                                                false, false);
}

// ---- CDNA5 async global->LDS copy (ASYNCcnt tracked) ----
// LDS flat aperture keeps the LDS byte offset in addr[31:0].
__device__ __forceinline__ unsigned int lds_off(const void* p) {
  return (unsigned int)(size_t)p;
}
__device__ __forceinline__ void async_g2l_b128(unsigned int ldsoff,
                                               const void* g) {
  asm volatile("global_load_async_to_lds_b128 %0, %1, off"
               :: "v"(ldsoff), "v"(g)
               : "memory");
}
__device__ __forceinline__ void async_wait0() {
  asm volatile("s_wait_asynccnt 0x0" ::: "memory");
}

// ---------------------------------------------------------------------------
// f32 -> f16 convert (strided rows, 256 cols). 4 elems / thread.
// ---------------------------------------------------------------------------
__global__ __launch_bounds__(256) void k_cvt_f16(const float* __restrict__ src,
                                                 int ldsrc,
                                                 _Float16* __restrict__ dst) {
  int i4 = (blockIdx.x * 256 + threadIdx.x) * 4;
  int r = i4 >> 8, c = i4 & 255;
  const float* s = src + (size_t)r * ldsrc + c;
  _Float16* d = dst + i4;
#pragma unroll
  for (int j = 0; j < 4; ++j) d[j] = (_Float16)s[j];
}

// ---------------------------------------------------------------------------
// GEMM: C[M,N] = (A[M,KA] @ W[N,KA]^T + bias) * outscale. A,W are f16.
// Tile 128x64, 256 threads, double-buffered async global->LDS staging.
// ---------------------------------------------------------------------------
template <bool OUT_F16>
__global__ __launch_bounds__(256) void k_gemm_nt(
    const _Float16* __restrict__ A, int lda, const _Float16* __restrict__ W,
    int ldw, const float* __restrict__ bias, void* __restrict__ Cout, int ldc,
    int KA, float outscale) {
  __shared__ __align__(16) _Float16 As[2][128 * 40];
  __shared__ __align__(16) _Float16 Ws[2][64 * 40];
  const int tid = threadIdx.x;
  const int wave = tid >> 5, lane = tid & 31;
  const int row = lane & 15, kh = lane >> 4;
  const int m0 = blockIdx.x * 128, n0 = blockIdx.y * 64;

  auto stage = [&](int buf, int k0) {
    // A tile: 128 rows x 4 chunks of 8 halves (16B each)
#pragma unroll
    for (int ch = tid; ch < 512; ch += 256) {
      int r = ch >> 2, c8 = (ch & 3) * 8;
      async_g2l_b128(lds_off(&As[buf][r * 40 + c8]),
                     &A[(size_t)(m0 + r) * lda + k0 + c8]);
    }
    // W tile: 64 rows x 4 chunks
    {
      int r = tid >> 2, c8 = (tid & 3) * 8;
      if (tid < 256)
        async_g2l_b128(lds_off(&Ws[buf][r * 40 + c8]),
                       &W[(size_t)(n0 + r) * ldw + k0 + c8]);
    }
  };

  v8f acc[4];
#pragma unroll
  for (int t = 0; t < 4; ++t) acc[t] = vzero8();

  stage(0, 0);
  async_wait0();
  __syncthreads();

  const int nk = KA >> 5;
  for (int ki = 0; ki < nk; ++ki) {
    const int cur = ki & 1;
    if (ki + 1 < nk) stage(cur ^ 1, (ki + 1) * 32);
    v16h af = frag_ld(&As[cur][(wave * 16 + row) * 40 + kh * 8]);
#pragma unroll
    for (int nt = 0; nt < 4; ++nt) {
      v16h bf = frag_ld(&Ws[cur][(nt * 16 + row) * 40 + kh * 8]);
      acc[nt] = wmma16(af, bf, acc[nt]);
    }
    async_wait0();
    __syncthreads();
  }

  const int cn = lane & 15, cm = (lane >> 4) * 8;
#pragma unroll
  for (int nt = 0; nt < 4; ++nt) {
    float bv = bias[n0 + nt * 16 + cn];
#pragma unroll
    for (int r = 0; r < 8; ++r) {
      float v = (acc[nt][r] + bv) * outscale;
      size_t idx = (size_t)(m0 + wave * 16 + cm + r) * ldc + n0 + nt * 16 + cn;
      if (OUT_F16)
        ((_Float16*)Cout)[idx] = (_Float16)v;
      else
        ((float*)Cout)[idx] = v;
    }
  }
}

// ---------------------------------------------------------------------------
// Standard MHA, one (b,h) per block, Lk = 128. Q pre-scaled (GEMM outscale).
// All operands f16. Async-staged Q/K; V staged transposed.
// ---------------------------------------------------------------------------
__global__ __launch_bounds__(256) void k_attn_std(
    const _Float16* __restrict__ Qh, const _Float16* __restrict__ Kh,
    const _Float16* __restrict__ Vh, _Float16* __restrict__ Ocat, int obase) {
  __shared__ __align__(16) _Float16 Qs[128 * 40];
  __shared__ __align__(16) _Float16 Ks[128 * 40];
  __shared__ __align__(16) _Float16 Vt[32 * 136];      // [d][k]
  __shared__ __align__(16) _Float16 Ps[8 * 16 * 136];  // per-wave prob strips
  const int tid = threadIdx.x, wave = tid >> 5, lane = tid & 31;
  const int row = lane & 15, kh = lane >> 4;
  const int h = blockIdx.x, b = blockIdx.y;
  const int cq = h * Dd;

#pragma unroll
  for (int ch = tid; ch < 512; ch += 256) {
    int r = ch >> 2, c8 = (ch & 3) * 8;
    size_t g = (size_t)(b * 128 + r) * Ee + cq + c8;
    async_g2l_b128(lds_off(&Qs[r * 40 + c8]), &Qh[g]);
    async_g2l_b128(lds_off(&Ks[r * 40 + c8]), &Kh[g]);
    v8h v = *(const v8h*)&Vh[g];
#pragma unroll
    for (int j = 0; j < 8; ++j) Vt[(c8 + j) * 136 + r] = v[j];
  }
  async_wait0();
  __syncthreads();

  v16h qf = frag_ld(&Qs[(wave * 16 + row) * 40 + kh * 8]);
  v8f s[8];
#pragma unroll
  for (int t = 0; t < 8; ++t) {
    v16h kf = frag_ld(&Ks[(t * 16 + row) * 40 + kh * 8]);
    s[t] = wmma16(qf, kf, vzero8());
  }
  // Softmax on WMMA C layout: row m = (lane>>4)*8+r, col = (lane&15)+16*t.
#pragma unroll
  for (int r = 0; r < 8; ++r) {
    float mx = -3.0e38f;
#pragma unroll
    for (int t = 0; t < 8; ++t) mx = fmaxf(mx, s[t][r]);
#pragma unroll
    for (int msk = 1; msk < 16; msk <<= 1)
      mx = fmaxf(mx, __shfl_xor(mx, msk, 32));
    float sum = 0.0f;
#pragma unroll
    for (int t = 0; t < 8; ++t) {
      float e = __expf(s[t][r] - mx);
      s[t][r] = e;
      sum += e;
    }
#pragma unroll
    for (int msk = 1; msk < 16; msk <<= 1) sum += __shfl_xor(sum, msk, 32);
    float inv = 1.0f / sum;
    int m = (lane >> 4) * 8 + r;
#pragma unroll
    for (int t = 0; t < 8; ++t)
      Ps[(wave * 16 + m) * 136 + t * 16 + (lane & 15)] =
          (_Float16)(s[t][r] * inv);
  }
  v8f o[2];
  o[0] = vzero8();
  o[1] = vzero8();
#pragma unroll
  for (int kc = 0; kc < 4; ++kc) {
    v16h pf = frag_ld(&Ps[(wave * 16 + row) * 136 + kc * 32 + kh * 8]);
#pragma unroll
    for (int nt = 0; nt < 2; ++nt) {
      v16h vf = frag_ld(&Vt[(nt * 16 + row) * 136 + kc * 32 + kh * 8]);
      o[nt] = wmma16(pf, vf, o[nt]);
    }
  }
  const int cn = lane & 15;
#pragma unroll
  for (int nt = 0; nt < 2; ++nt)
#pragma unroll
    for (int r = 0; r < 8; ++r) {
      int m = (lane >> 4) * 8 + r;
      Ocat[(size_t)(b * 128 + wave * 16 + m) * (4 * Ee) + obase + cq +
           nt * 16 + cn] = (_Float16)o[nt][r];
    }
}

// ---------------------------------------------------------------------------
// du-branch attention (per-position keys), rank-2 folded. Lk = 64.
// S = Qs@Kb^T + CI0*a0[l] + CI1*a1[l]; softmax; O = P@Vb + c0*uv0h + c1*uv1h.
// ---------------------------------------------------------------------------
__global__ __launch_bounds__(256) void k_attn_du(
    const _Float16* __restrict__ Qh, const _Float16* __restrict__ Kb,
    const _Float16* __restrict__ Vb, const float* __restrict__ CI,
    const float* __restrict__ uk0, const float* __restrict__ uk1,
    const float* __restrict__ uv0, const float* __restrict__ uv1,
    _Float16* __restrict__ Ocat) {
  __shared__ __align__(16) _Float16 Qs[128 * 40];
  __shared__ __align__(16) _Float16 Ks[64 * 40];
  __shared__ __align__(16) _Float16 Vt[32 * 72];
  __shared__ __align__(16) _Float16 Ps[8 * 16 * 72];
  __shared__ float a0s[128], a1s[128], c0s[128], c1s[128];
  __shared__ float ukh0[32], ukh1[32], uvh0[32], uvh1[32];
  const int tid = threadIdx.x, wave = tid >> 5, lane = tid & 31;
  const int row = lane & 15, kh = lane >> 4;
  const int h = blockIdx.x, b = blockIdx.y;
  const int cq = h * Dd;

#pragma unroll
  for (int ch = tid; ch < 512; ch += 256) {
    int r = ch >> 2, c8 = (ch & 3) * 8;
    async_g2l_b128(lds_off(&Qs[r * 40 + c8]),
                   &Qh[(size_t)(b * 128 + r) * Ee + cq + c8]);
  }
  {
    int r = tid >> 2, c8 = (tid & 3) * 8;  // 64 rows x 4 chunks = 256 threads
    size_t g = (size_t)(b * 64 + r) * Ee + cq + c8;
    async_g2l_b128(lds_off(&Ks[r * 40 + c8]), &Kb[g]);
    v8h v = *(const v8h*)&Vb[g];
#pragma unroll
    for (int j = 0; j < 8; ++j) Vt[(c8 + j) * 72 + r] = v[j];
  }
  if (tid < 32) {
    ukh0[tid] = uk0[cq + tid];
    ukh1[tid] = uk1[cq + tid];
    uvh0[tid] = uv0[cq + tid];
    uvh1[tid] = uv1[cq + tid];
  }
  async_wait0();
  __syncthreads();
  if (tid < 128) {  // a0/a1 per query row (Q already scaled)
    float d0 = 0.f, d1 = 0.f;
#pragma unroll 4
    for (int c = 0; c < 32; ++c) {
      float q = (float)Qs[tid * 40 + c];
      d0 = fmaf(q, ukh0[c], d0);
      d1 = fmaf(q, ukh1[c], d1);
    }
    a0s[tid] = d0;
    a1s[tid] = d1;
  }
  __syncthreads();

  v16h qf = frag_ld(&Qs[(wave * 16 + row) * 40 + kh * 8]);
  v8f s[4];
#pragma unroll
  for (int t = 0; t < 4; ++t) {
    v16h kf = frag_ld(&Ks[(t * 16 + row) * 40 + kh * 8]);
    s[t] = wmma16(qf, kf, vzero8());
  }

#pragma unroll
  for (int r = 0; r < 8; ++r) {
    int m = (lane >> 4) * 8 + r;
    int l = wave * 16 + m;
    float ci0v[4], ci1v[4];
    float mx = -3.0e38f;
#pragma unroll
    for (int t = 0; t < 4; ++t) {
      size_t cidx = ((size_t)(b * Ll + l) * Kk + t * 16 + (lane & 15)) * 2;
      ci0v[t] = CI[cidx];
      ci1v[t] = CI[cidx + 1];
      s[t][r] += ci0v[t] * a0s[l] + ci1v[t] * a1s[l];
      mx = fmaxf(mx, s[t][r]);
    }
#pragma unroll
    for (int msk = 1; msk < 16; msk <<= 1)
      mx = fmaxf(mx, __shfl_xor(mx, msk, 32));
    float sum = 0.0f;
#pragma unroll
    for (int t = 0; t < 4; ++t) {
      float e = __expf(s[t][r] - mx);
      s[t][r] = e;
      sum += e;
    }
#pragma unroll
    for (int msk = 1; msk < 16; msk <<= 1) sum += __shfl_xor(sum, msk, 32);
    float inv = 1.0f / sum;
    float cc0 = 0.f, cc1 = 0.f;
#pragma unroll
    for (int t = 0; t < 4; ++t) {
      float p = s[t][r] * inv;
      Ps[(wave * 16 + m) * 72 + t * 16 + (lane & 15)] = (_Float16)p;
      cc0 = fmaf(p, ci0v[t], cc0);
      cc1 = fmaf(p, ci1v[t], cc1);
    }
#pragma unroll
    for (int msk = 1; msk < 16; msk <<= 1) {
      cc0 += __shfl_xor(cc0, msk, 32);
      cc1 += __shfl_xor(cc1, msk, 32);
    }
    if ((lane & 15) == 0) {
      c0s[l] = cc0;
      c1s[l] = cc1;
    }
  }

  v8f o[2];
  o[0] = vzero8();
  o[1] = vzero8();
#pragma unroll
  for (int kc = 0; kc < 2; ++kc) {
    v16h pf = frag_ld(&Ps[(wave * 16 + row) * 72 + kc * 32 + kh * 8]);
#pragma unroll
    for (int nt = 0; nt < 2; ++nt) {
      v16h vf = frag_ld(&Vt[(nt * 16 + row) * 72 + kc * 32 + kh * 8]);
      o[nt] = wmma16(pf, vf, o[nt]);
    }
  }
  const int cn = lane & 15;
#pragma unroll
  for (int nt = 0; nt < 2; ++nt)
#pragma unroll
    for (int r = 0; r < 8; ++r) {
      int m = (lane >> 4) * 8 + r;
      int l = wave * 16 + m;
      int d = nt * 16 + cn;
      float val = o[nt][r] + c0s[l] * uvh0[d] + c1s[l] * uvh1[d];
      Ocat[(size_t)(b * 128 + l) * (4 * Ee) + cq + d] = (_Float16)val;
    }
}

// ---------------------------------------------------------------------------
// Wcat[e, br*E+j] = sum_m dim_w[e, br*E+m] * Wo_br[m, j]  (written as f16)
// ---------------------------------------------------------------------------
__global__ __launch_bounds__(256) void k_prep_wcat(
    const float* __restrict__ dim_w, const float* __restrict__ ow0,
    const float* __restrict__ ow1, const float* __restrict__ ow2,
    const float* __restrict__ ow3, _Float16* __restrict__ Wcat) {
  const int e = blockIdx.x, br = blockIdx.y, j = threadIdx.x;
  const float* ow = (br == 0) ? ow0 : (br == 1) ? ow1 : (br == 2) ? ow2 : ow3;
  const float* dwr = dim_w + (size_t)e * (4 * Ee) + br * Ee;
  float acc = 0.0f;
  for (int m = 0; m < Ee; ++m) acc = fmaf(dwr[m], ow[(size_t)m * Ee + j], acc);
  Wcat[(size_t)e * (4 * Ee) + br * Ee + j] = (_Float16)acc;
}

// bc = dim_b + sum_br dim_w_br @ bo_br ; uk/uv rank-2 vectors for du branch.
__global__ __launch_bounds__(256) void k_prep_small(
    const float* __restrict__ dim_w, const float* __restrict__ dim_b,
    const float* __restrict__ bo0, const float* __restrict__ bo1,
    const float* __restrict__ bo2, const float* __restrict__ bo3,
    const float* __restrict__ du_in_w, const float* __restrict__ ci_w,
    float* __restrict__ bc, float* __restrict__ uk0, float* __restrict__ uk1,
    float* __restrict__ uv0, float* __restrict__ uv1) {
  const int t = threadIdx.x;
  const float* bos[4] = {bo0, bo1, bo2, bo3};
  float acc = dim_b[t];
  for (int br = 0; br < 4; ++br) {
    const float* dwr = dim_w + (size_t)t * (4 * Ee) + br * Ee;
    const float* bo = bos[br];
    for (int m = 0; m < Ee; ++m) acc = fmaf(dwr[m], bo[m], acc);
  }
  bc[t] = acc;
  float k0 = 0.f, k1 = 0.f, w0 = 0.f, w1 = 0.f;
  for (int e = 0; e < Ee; ++e) {
    float wk = du_in_w[(size_t)(Ee + t) * Ee + e];
    float wv = du_in_w[(size_t)(2 * Ee + t) * Ee + e];
    float c0 = ci_w[(size_t)e * (Ee + 2) + Ee];
    float c1 = ci_w[(size_t)e * (Ee + 2) + Ee + 1];
    k0 = fmaf(wk, c0, k0);
    k1 = fmaf(wk, c1, k1);
    w0 = fmaf(wv, c0, w0);
    w1 = fmaf(wv, c1, w1);
  }
  uk0[t] = k0;
  uk1[t] = k1;
  uv0[t] = w0;
  uv1[t] = w1;
}

// ---------------------------------------------------------------------------
// BatchNorm (training stats over B*L rows per channel) + ReLU + residual.
// ---------------------------------------------------------------------------
__global__ __launch_bounds__(256) void k_bn_stats(const float* __restrict__ x,
                                                  float* __restrict__ meanv,
                                                  float* __restrict__ rstdv) {
  const int e = blockIdx.x, tid = threadIdx.x;
  float s = 0.f, s2 = 0.f;
  for (int r = tid; r < ROWS; r += 256) {
    float v = x[(size_t)r * Ee + e];
    s += v;
    s2 = fmaf(v, v, s2);
  }
#pragma unroll
  for (int msk = 1; msk < 32; msk <<= 1) {
    s += __shfl_xor(s, msk, 32);
    s2 += __shfl_xor(s2, msk, 32);
  }
  __shared__ float ps[8], ps2[8];
  if ((tid & 31) == 0) {
    ps[tid >> 5] = s;
    ps2[tid >> 5] = s2;
  }
  __syncthreads();
  if (tid == 0) {
    float S = 0.f, S2 = 0.f;
    for (int i = 0; i < 8; ++i) {
      S += ps[i];
      S2 += ps2[i];
    }
    float m = S * (1.0f / ROWS);
    float var = S2 * (1.0f / ROWS) - m * m;
    meanv[e] = m;
    rstdv[e] = rsqrtf(var + 1e-5f);
  }
}

__global__ __launch_bounds__(256) void k_bn_apply(
    const float* __restrict__ x, const float* __restrict__ meanv,
    const float* __restrict__ rstdv, const float* __restrict__ g,
    const float* __restrict__ bb, const float* __restrict__ UU,
    float* __restrict__ out) {
  int idx = blockIdx.x * 256 + threadIdx.x;
  int e = idx & (Ee - 1);
  float y = g[e] * (x[idx] - meanv[e]) * rstdv[e] + bb[e];
  y = fmaxf(y, 0.0f);
  out[idx] = y + UU[idx];
}

// ---------------------------------------------------------------------------
extern "C" void kernel_launch(void* const* d_in, const int* in_sizes, int n_in,
                              void* d_out, int out_size, void* d_ws,
                              size_t ws_size, hipStream_t stream) {
  (void)in_sizes; (void)n_in; (void)out_size; (void)ws_size;
  const float* UU = (const float*)d_in[0];
  const float* DU = (const float*)d_in[1];
  const float* INm = (const float*)d_in[2];
  const float* TA = (const float*)d_in[3];
  const float* CI = (const float*)d_in[4];
  const float* ta_in_w = (const float*)d_in[5];
  const float* ta_in_b = (const float*)d_in[6];
  const float* ta_out_w = (const float*)d_in[7];
  const float* ta_out_b = (const float*)d_in[8];
  const float* uu_in_w = (const float*)d_in[9];
  const float* uu_in_b = (const float*)d_in[10];
  const float* uu_out_w = (const float*)d_in[11];
  const float* uu_out_b = (const float*)d_in[12];
  const float* du_in_w = (const float*)d_in[13];
  const float* du_in_b = (const float*)d_in[14];
  const float* du_out_w = (const float*)d_in[15];
  const float* du_out_b = (const float*)d_in[16];
  const float* in_in_w = (const float*)d_in[17];
  const float* in_in_b = (const float*)d_in[18];
  const float* in_out_w = (const float*)d_in[19];
  const float* in_out_b = (const float*)d_in[20];
  const float* ci_w = (const float*)d_in[21];
  const float* ci_b = (const float*)d_in[22];
  const float* dim_w = (const float*)d_in[23];
  const float* dim_b = (const float*)d_in[24];
  const float* bn_g = (const float*)d_in[25];
  const float* bn_b = (const float*)d_in[26];

  // ---- workspace layout: f16 pool then f32 pool ----
  _Float16* hws = (_Float16*)d_ws;
  size_t hoff = 0;
  auto halloc = [&](size_t n) { _Float16* p = hws + hoff; hoff += n; return p; };
  const size_t SZ_BL = (size_t)ROWS * Ee;      // 2048*256
  const size_t SZ_BK = (size_t)Bb * Kk * Ee;   // 1024*256
  _Float16* uu16 = halloc(SZ_BL);
  _Float16* ta16 = halloc(SZ_BL);
  _Float16* in16 = halloc(SZ_BL);
  _Float16* du16 = halloc(SZ_BK);
  _Float16* taw16 = halloc((size_t)3 * Ee * Ee);
  _Float16* uuw16 = halloc((size_t)3 * Ee * Ee);
  _Float16* duw16 = halloc((size_t)3 * Ee * Ee);
  _Float16* inw16 = halloc((size_t)3 * Ee * Ee);
  _Float16* ciw16 = halloc((size_t)Ee * Ee);
  _Float16* qp_du = halloc(SZ_BL);
  _Float16* qp_ta = halloc(SZ_BL);
  _Float16* qp_uu = halloc(SZ_BL);
  _Float16* qp_in = halloc(SZ_BL);
  _Float16* kp_ta = halloc(SZ_BL);
  _Float16* vp_ta = halloc(SZ_BL);
  _Float16* kp_uu = halloc(SZ_BL);
  _Float16* vp_uu = halloc(SZ_BL);
  _Float16* kp_in = halloc(SZ_BL);
  _Float16* vp_in = halloc(SZ_BL);
  _Float16* dub16 = halloc(SZ_BK);
  _Float16* kbase = halloc(SZ_BK);
  _Float16* vbase = halloc(SZ_BK);
  _Float16* ocat = halloc((size_t)ROWS * 4 * Ee);
  _Float16* wcat = halloc((size_t)Ee * 4 * Ee);
  float* fws = (float*)(hws + hoff);
  size_t foff = 0;
  auto falloc = [&](size_t n) { float* p = fws + foff; foff += n; return p; };
  float* bc = falloc(Ee);
  float* uk0 = falloc(Ee);
  float* uk1 = falloc(Ee);
  float* uv0 = falloc(Ee);
  float* uv1 = falloc(Ee);
  float* xbuf = falloc(SZ_BL);
  float* meanv = falloc(Ee);
  float* rstdv = falloc(Ee);

  dim3 blk(256);
  // ---- f32 -> f16 conversions (activations + weights) ----
  k_cvt_f16<<<dim3(512), blk, 0, stream>>>(UU, Ee, uu16);
  k_cvt_f16<<<dim3(512), blk, 0, stream>>>(TA, Ee, ta16);
  k_cvt_f16<<<dim3(512), blk, 0, stream>>>(INm, Ee, in16);
  k_cvt_f16<<<dim3(256), blk, 0, stream>>>(DU, Ee, du16);
  k_cvt_f16<<<dim3(192), blk, 0, stream>>>(ta_in_w, Ee, taw16);
  k_cvt_f16<<<dim3(192), blk, 0, stream>>>(uu_in_w, Ee, uuw16);
  k_cvt_f16<<<dim3(192), blk, 0, stream>>>(du_in_w, Ee, duw16);
  k_cvt_f16<<<dim3(192), blk, 0, stream>>>(in_in_w, Ee, inw16);
  k_cvt_f16<<<dim3(64), blk, 0, stream>>>(ci_w, Ee + 2, ciw16);
  // ---- per-call weight folding ----
  k_prep_wcat<<<dim3(Ee, 4), blk, 0, stream>>>(dim_w, du_out_w, ta_out_w,
                                               uu_out_w, in_out_w, wcat);
  k_prep_small<<<1, blk, 0, stream>>>(dim_w, dim_b, du_out_b, ta_out_b,
                                      uu_out_b, in_out_b, du_in_w, ci_w, bc,
                                      uk0, uk1, uv0, uv1);
  // ---- Q projections (SCALE folded into output) ----
  dim3 g2048(16, 4), g1024(8, 4);
  k_gemm_nt<true><<<g2048, blk, 0, stream>>>(uu16, Ee, duw16, Ee, du_in_b,
                                             qp_du, Ee, Ee, ATT_SCALE);
  k_gemm_nt<true><<<g2048, blk, 0, stream>>>(uu16, Ee, taw16, Ee, ta_in_b,
                                             qp_ta, Ee, Ee, ATT_SCALE);
  k_gemm_nt<true><<<g2048, blk, 0, stream>>>(uu16, Ee, uuw16, Ee, uu_in_b,
                                             qp_uu, Ee, Ee, ATT_SCALE);
  k_gemm_nt<true><<<g2048, blk, 0, stream>>>(uu16, Ee, inw16, Ee, in_in_b,
                                             qp_in, Ee, Ee, ATT_SCALE);
  // ---- K/V projections for std branches ----
  k_gemm_nt<true><<<g2048, blk, 0, stream>>>(ta16, Ee, taw16 + Ee * Ee, Ee,
                                             ta_in_b + Ee, kp_ta, Ee, Ee, 1.f);
  k_gemm_nt<true><<<g2048, blk, 0, stream>>>(ta16, Ee, taw16 + 2 * Ee * Ee, Ee,
                                             ta_in_b + 2 * Ee, vp_ta, Ee, Ee,
                                             1.f);
  k_gemm_nt<true><<<g2048, blk, 0, stream>>>(uu16, Ee, uuw16 + Ee * Ee, Ee,
                                             uu_in_b + Ee, kp_uu, Ee, Ee, 1.f);
  k_gemm_nt<true><<<g2048, blk, 0, stream>>>(uu16, Ee, uuw16 + 2 * Ee * Ee, Ee,
                                             uu_in_b + 2 * Ee, vp_uu, Ee, Ee,
                                             1.f);
  k_gemm_nt<true><<<g2048, blk, 0, stream>>>(in16, Ee, inw16 + Ee * Ee, Ee,
                                             in_in_b + Ee, kp_in, Ee, Ee, 1.f);
  k_gemm_nt<true><<<g2048, blk, 0, stream>>>(in16, Ee, inw16 + 2 * Ee * Ee, Ee,
                                             in_in_b + 2 * Ee, vp_in, Ee, Ee,
                                             1.f);
  // ---- du branch l-independent part ----
  k_gemm_nt<true><<<g1024, blk, 0, stream>>>(du16, Ee, ciw16, Ee, ci_b, dub16,
                                             Ee, Ee, 1.f);
  k_gemm_nt<true><<<g1024, blk, 0, stream>>>(dub16, Ee, duw16 + Ee * Ee, Ee,
                                             du_in_b + Ee, kbase, Ee, Ee, 1.f);
  k_gemm_nt<true><<<g1024, blk, 0, stream>>>(dub16, Ee, duw16 + 2 * Ee * Ee, Ee,
                                             du_in_b + 2 * Ee, vbase, Ee, Ee,
                                             1.f);
  // ---- attention (o_cat branch order = [du, ta, uu, in]) ----
  dim3 ga(Hh, Bb);
  k_attn_du<<<ga, blk, 0, stream>>>(qp_du, kbase, vbase, CI, uk0, uk1, uv0,
                                    uv1, ocat);
  k_attn_std<<<ga, blk, 0, stream>>>(qp_ta, kp_ta, vp_ta, ocat, 1 * Ee);
  k_attn_std<<<ga, blk, 0, stream>>>(qp_uu, kp_uu, vp_uu, ocat, 2 * Ee);
  k_attn_std<<<ga, blk, 0, stream>>>(qp_in, kp_in, vp_in, ocat, 3 * Ee);
  // ---- fused out-projections + dim: x = o_cat @ Wcat^T + bc (f32 out) ----
  k_gemm_nt<false><<<dim3(16, 4), blk, 0, stream>>>(ocat, 4 * Ee, wcat, 4 * Ee,
                                                    bc, xbuf, Ee, 4 * Ee, 1.f);
  // ---- BN + ReLU + residual ----
  k_bn_stats<<<dim3(Ee), blk, 0, stream>>>(xbuf, meanv, rstdv);
  k_bn_apply<<<dim3(ROWS * Ee / 256), blk, 0, stream>>>(
      xbuf, meanv, rstdv, bn_g, bn_b, UU, (float*)d_out);
}
